// PIPNet_40183714021718
// MI455X (gfx1250) — matrix operand order repaired and checked by reading, not compiled, not run on hardware
//
#include <hip/hip_runtime.h>
#include <hip/hip_bf16.h>

typedef _Float16 half_t;
typedef __attribute__((ext_vector_type(16))) _Float16 v16h;
typedef __attribute__((ext_vector_type(8)))  _Float16 v8h;
typedef __attribute__((ext_vector_type(2)))  _Float16 v2h;
typedef __attribute__((ext_vector_type(8)))  float    v8f;

#define B_   8
#define L_   2048
#define V_   8192
#define C_   64
#define F_   130          // 2*(C+1)
#define LM   64           // L rows per block
#define KV   128          // verts per K-chunk (4 WMMA K-steps)
#define WSTR 136          // padded LDS row stride in halves (272B, 16B-aligned)

// ---------------------------------------------------------------------------
// Kernel 1: RBF weights + WMMA matmul S = w @ feats, dens = row-sum(w).
// Writes X[B*L, 130]: cols [side*65 .. +63] = S/dens, col side*65+64 = dens.
// grid.x = B*2*(L/64) = 512, block = 256 (8 waves).
// Wave w: col-tile ct = w&3, row-tiles rt = {(w>>2)*2, (w>>2)*2+1}.
// Per 128-K chunk: 4 K-steps x 2 row-tiles = 8 v_wmma per wave, B-frag reused,
// 2 barriers per chunk, verts double-buffered, next-chunk feats prefetched.
// ---------------------------------------------------------------------------
__global__ __launch_bounds__(256) void rbf_wmma_kernel(
    const float* __restrict__ locsL, const float* __restrict__ locsR,
    const float* __restrict__ vertsL, const float* __restrict__ vertsR,
    const float* __restrict__ featsL, const float* __restrict__ featsR,
    float* __restrict__ X)
{
    __shared__ __align__(16) float  locs_s[LM * 3];
    __shared__ __align__(16) float  verts_s[2][KV * 3];
    __shared__ __align__(16) half_t w_s[LM * WSTR];    // w tile  [row m][k]
    __shared__ __align__(16) half_t f_s[C_ * WSTR];    // featsT  [chan c][k]
    __shared__ __align__(16) float  dens_part[LM * 4];
    __shared__ __align__(16) float  dens_s[LM];

    const int t     = threadIdx.x;
    const int ltile = blockIdx.x & 31;        // L/64 = 32
    const int side  = (blockIdx.x >> 5) & 1;
    const int b     = blockIdx.x >> 6;

    const float* locs  = side ? locsR  : locsL;
    const float* verts = side ? vertsR : vertsL;
    const float* feats = side ? featsR : featsL;

    const float* feats_b = feats + (size_t)b * V_ * C_;
    const float* verts_b = verts + (size_t)b * V_ * 3;

    const int l0 = ltile * LM;
    if (t < LM * 3) locs_s[t] = locs[(size_t)b * L_ * 3 + l0 * 3 + t];
    // prologue: stage verts for chunk 0 into buffer 0
    for (int i = t; i < KV * 3; i += 256) verts_s[0][i] = verts_b[i];
    __syncthreads();

    // w-tile assignment: row m = t&63, k-group kg = t>>6 (32 k's each)
    const int m  = t & 63;
    const int kg = t >> 6;
    const float lx = locs_s[m * 3 + 0];
    const float ly = locs_s[m * 3 + 1];
    const float lz = locs_s[m * 3 + 2];

    // wave tiling
    const int wave = t >> 5;
    const int lane = t & 31;
    const int ct   = wave & 3;
    const int rtb  = (wave >> 2) * 2;         // first of two row-tiles
    const int lmod = lane & 15;
    const int g    = lane >> 4;

    v8f acc0 = {};
    v8f acc1 = {};
    float dens_acc = 0.0f;

    int buf = 0;
    for (int v0 = 0; v0 < V_; v0 += KV) {
        const float* vcur = verts_s[buf];

        // ---- stage feats chunk transposed (f32 -> packed f16 half2 stores)
        //      128 rows x 16 float4, row-pairs: 1024 pairs / 256 thr = 4 each
#pragma unroll
        for (int q = 0; q < 4; ++q) {
            int pi = t + q * 256;
            int r  = (pi >> 4) * 2;           // even vert row in chunk
            int c4 = pi & 15;                 // float4 within channel row
            float4 fa = ((const float4*)(feats_b + (size_t)(v0 + r) * C_))[c4];
            float4 fb = ((const float4*)(feats_b + (size_t)(v0 + r + 1) * C_))[c4];
            *(v2h*)&f_s[(c4 * 4 + 0) * WSTR + r] = (v2h){(half_t)fa.x, (half_t)fb.x};
            *(v2h*)&f_s[(c4 * 4 + 1) * WSTR + r] = (v2h){(half_t)fa.y, (half_t)fb.y};
            *(v2h*)&f_s[(c4 * 4 + 2) * WSTR + r] = (v2h){(half_t)fa.z, (half_t)fb.z};
            *(v2h*)&f_s[(c4 * 4 + 3) * WSTR + r] = (v2h){(half_t)fa.w, (half_t)fb.w};
        }

        // ---- stage next chunk's verts into the other buffer ----
        if (v0 + KV < V_) {
            const float* vnext = verts_b + (size_t)(v0 + KV) * 3;
            for (int i = t; i < KV * 3; i += 256) verts_s[buf ^ 1][i] = vnext[i];
            // prefetch next chunk's feats (32KB): 256 thr x 128B
            __builtin_prefetch(feats_b + (size_t)(v0 + KV) * C_ + t * 32, 0, 1);
        }

        // ---- compute 64x128 w tile (32 exps/thread), overlapped with staging
#pragma unroll
        for (int j = 0; j < 32; ++j) {
            int k = kg * 32 + j;
            float dx = lx - vcur[k * 3 + 0];
            float dy = ly - vcur[k * 3 + 1];
            float dz = lz - vcur[k * 3 + 2];
            float d  = sqrtf(dx * dx + dy * dy + dz * dz);
            float wv = __expf(d * (-1.0f / 2.5f));
            dens_acc += wv;
            w_s[m * WSTR + k] = (half_t)wv;
        }
        __syncthreads();

        // ---- fragments (contiguous k = ks+16*(lane/16)+i, same in A and B) + 8 WMMAs, B-frag reused ----
#pragma unroll
        for (int ks = 0; ks < KV; ks += 32) {
            // B: 32x16 f16. lane col n = lane%16, half i -> K = ks+(lane/16)*16+i
            const int bc = ct * 16 + lmod;
            const int kb = ks + g * 16;
            v8h b0 = *(const v8h*)&f_s[bc * WSTR + kb];
            v8h b1 = *(const v8h*)&f_s[bc * WSTR + kb + 8];
            v16h bf = __builtin_shufflevector(b0, b1,
                      0,1,2,3,4,5,6,7,8,9,10,11,12,13,14,15);
            // A row-tile 0
            {
                const int arow = (rtb + 0) * 16 + lmod;
                v8h a0 = *(const v8h*)&w_s[arow * WSTR + ks + g * 16];
                v8h a1 = *(const v8h*)&w_s[arow * WSTR + ks + g * 16 + 8];
                v16h a = __builtin_shufflevector(a0, a1,
                         0,1,2,3,4,5,6,7,8,9,10,11,12,13,14,15);
                acc0 = __builtin_amdgcn_wmma_f32_16x16x32_f16(
                           false, a, false, bf, (short)0, acc0, false, false);
            }
            // A row-tile 1
            {
                const int arow = (rtb + 1) * 16 + lmod;
                v8h a0 = *(const v8h*)&w_s[arow * WSTR + ks + g * 16];
                v8h a1 = *(const v8h*)&w_s[arow * WSTR + ks + g * 16 + 8];
                v16h a = __builtin_shufflevector(a0, a1,
                         0,1,2,3,4,5,6,7,8,9,10,11,12,13,14,15);
                acc1 = __builtin_amdgcn_wmma_f32_16x16x32_f16(
                           false, a, false, bf, (short)0, acc1, false, false);
            }
        }
        __syncthreads();
        buf ^= 1;
    }

    // ---- reduce dens over k-groups ----
    dens_part[m * 4 + kg] = dens_acc;
    __syncthreads();
    if (t < LM) {
        float s = 0.0f;
#pragma unroll
        for (int q = 0; q < 4; ++q) s += dens_part[t * 4 + q];
        dens_s[t] = s;
    }
    __syncthreads();

    // ---- write S/dens tiles: D layout VGPR r -> row rt*16 + g*8 + r ----
    {
        const int c = ct * 16 + lmod;
#pragma unroll
        for (int r = 0; r < 8; ++r) {
            int row0 = (rtb + 0) * 16 + g * 8 + r;
            int row1 = (rtb + 1) * 16 + g * 8 + r;
            X[(size_t)(b * L_ + l0 + row0) * F_ + side * 65 + c] = acc0[r] / dens_s[row0];
            X[(size_t)(b * L_ + l0 + row1) * F_ + side * 65 + c] = acc1[r] / dens_s[row1];
        }
    }
    if (t < LM) {
        X[(size_t)(b * L_ + l0 + t) * F_ + side * 65 + 64] = dens_s[t];
    }
}

// ---------------------------------------------------------------------------
// Kernel 2: tiny MLP head: out = relu(x@w1^T + b1) @ w2^T + b2, per row.
// grid = 64 x 256 = 16384 = B*L rows. w1 row loads are wave-uniform (scalar).
// ---------------------------------------------------------------------------
__global__ __launch_bounds__(256) void mlp_head_kernel(
    const float* __restrict__ X,
    const float* __restrict__ w1, const float* __restrict__ b1,
    const float* __restrict__ w2, const float* __restrict__ b2,
    float* __restrict__ out)
{
    const int row = blockIdx.x * 256 + threadIdx.x;
    const float* xr = X + (size_t)row * F_;
    float x[F_];
#pragma unroll
    for (int i = 0; i < F_; ++i) x[i] = xr[i];

    float o = b2[0];
    for (int j = 0; j < F_; ++j) {
        const float* wr = w1 + j * F_;
        float h = b1[j];
#pragma unroll
        for (int i = 0; i < F_; ++i) h = fmaf(x[i], wr[i], h);
        o = fmaf(fmaxf(h, 0.0f), w2[j], o);
    }
    out[row] = o;
}

extern "C" void kernel_launch(void* const* d_in, const int* in_sizes, int n_in,
                              void* d_out, int out_size, void* d_ws, size_t ws_size,
                              hipStream_t stream) {
    const float* locsL  = (const float*)d_in[0];
    const float* locsR  = (const float*)d_in[1];
    const float* vertsL = (const float*)d_in[2];
    const float* vertsR = (const float*)d_in[3];
    const float* featsL = (const float*)d_in[4];
    const float* featsR = (const float*)d_in[5];
    const float* w1     = (const float*)d_in[6];
    const float* b1     = (const float*)d_in[7];
    const float* w2     = (const float*)d_in[8];
    const float* b2     = (const float*)d_in[9];

    float* X = (float*)d_ws;   // [B*L, 130] = 8.5 MB scratch

    // B * 2 sides * (L/64) = 512 blocks
    rbf_wmma_kernel<<<B_ * 2 * (L_ / LM), 256, 0, stream>>>(
        locsL, locsR, vertsL, vertsR, featsL, featsR, X);

    mlp_head_kernel<<<(B_ * L_) / 256, 256, 0, stream>>>(
        X, w1, b1, w2, b2, (float*)d_out);
}
